// EncoderLayer_2723009266089
// MI455X (gfx1250) — matrix operand order repaired and checked by reading, not compiled, not run on hardware
//
#include <hip/hip_runtime.h>
#include <hip/hip_bf16.h>

// ---------------------------------------------------------------------------
// CDNA5 (gfx1250) transformer encoder layer, round 3.
// - All GEMMs + both attention matmuls on v_wmma_f32_16x16x32_bf16.
// - QKV projections emit bf16 so attention Q/K/V tiles are raw copies ->
//   GLOBAL_LOAD_ASYNC_TO_LDS_B128 (ASYNCcnt) via the gfx1250 builtin.
// - Double-buffered LDS pipelines HBM->LDS staging against WMMA.
// ---------------------------------------------------------------------------

typedef __attribute__((ext_vector_type(16))) __bf16          v16bf;
typedef __attribute__((ext_vector_type(16))) unsigned short  v16u;
typedef __attribute__((ext_vector_type(8)))  float           v8f;

#define WMMA_BF16(a, b, c) \
    __builtin_amdgcn_wmma_f32_16x16x32_bf16(false, (a), false, (b), (short)0, (c), false, false)

// ---- async global->LDS path (guarded; falls back to load+ds_store) --------
#if defined(__has_builtin)
#  if __has_builtin(__builtin_amdgcn_global_load_async_to_lds_b128) && \
      __has_builtin(__builtin_amdgcn_s_wait_asynccnt)
#    define ASYNC_LDS 1
#  endif
#endif
#ifndef ASYNC_LDS
#  define ASYNC_LDS 0
#endif

#if ASYNC_LDS
// builtin signature: (int4 AS1*, int4 AS3*, Imm int offset, Imm int cpol)
typedef int v4i __attribute__((vector_size(16)));
typedef __attribute__((address_space(1))) v4i gv4i;
typedef __attribute__((address_space(3))) v4i lv4i;
#endif

// copy 16 bytes global -> LDS (per active lane)
__device__ __forceinline__ void cp_g2l_b128(const unsigned short* g, unsigned short* l) {
#if ASYNC_LDS
    __builtin_amdgcn_global_load_async_to_lds_b128(
        (gv4i*)(unsigned long long)(const void*)g,
        (lv4i*)(unsigned int)(unsigned long long)(void*)l,
        0, 0);
#else
    *(uint4*)l = *(const uint4*)g;
#endif
}

__device__ __forceinline__ void async_wait() {
#if ASYNC_LDS
    __builtin_amdgcn_s_wait_asynccnt(0);
#endif
}

__device__ __forceinline__ unsigned short f32_to_bf16(float f) {
    union { float f; unsigned int u; } x; x.f = f;
    unsigned int r = x.u + 0x7FFFu + ((x.u >> 16) & 1u);   // RNE
    return (unsigned short)(r >> 16);
}

// A-fragment (16x32, 16-bit): lane = M row (lane%16); element e -> K per ISA:
// e<8 -> 8*half+e ; e>=8 -> 16+8*half+(e-8), half = lane/16.
__device__ __forceinline__ v16bf load_a_frag(const unsigned short* base, int ld) {
    int lane = threadIdx.x & 31;
    int row  = lane & 15;
    int half = lane >> 4;
    v16u u;
#pragma unroll
    for (int e = 0; e < 16; ++e) {
        int k = (e < 8) ? (8 * half + e) : (16 + 8 * half + (e - 8));
        u[e] = base[row * ld + k];
    }
    return __builtin_bit_cast(v16bf, u);
}

// B-fragment (32x16, 16-bit): lane%16 = N column, element e -> K as for A.
__device__ __forceinline__ v16bf load_b_frag(const unsigned short* base, int ld) {
    int lane = threadIdx.x & 31;
    v16u u;
#pragma unroll
    for (int e = 0; e < 16; ++e) u[e] = base[((e < 8) ? (8 * (lane >> 4) + e) : (16 + 8 * (lane >> 4) + (e - 8))) * ld + (lane & 15)];
    return __builtin_bit_cast(v16bf, u);
}

// B-fragment of a transposed LDS tile (K^T): lane%16 walks rows (N), e walks columns (K as for A).
__device__ __forceinline__ v16bf load_bT_frag(const unsigned short* base, int ld) {
    int lane = threadIdx.x & 31;
    v16u u;
#pragma unroll
    for (int e = 0; e < 16; ++e) u[e] = base[(lane & 15) * ld + ((e < 8) ? (8 * (lane >> 4) + e) : (16 + 8 * (lane >> 4) + (e - 8)))];
    return __builtin_bit_cast(v16bf, u);
}

// ---------------------------------------------------------------------------
// fp32 -> bf16 weight conversion (one-time)
// ---------------------------------------------------------------------------
__global__ void cvt_bf16_kernel(const float* __restrict__ in,
                                unsigned short* __restrict__ out, int n) {
    int i = blockIdx.x * blockDim.x + threadIdx.x;
    if (i < n) out[i] = f32_to_bf16(in[i]);
}

// ---------------------------------------------------------------------------
// C[M,N] = A[M,K] (f32 -> bf16) @ W[K,N] (bf16) + bias, opt ReLU, opt bf16 out.
// 256 threads = 8 waves (4M x 2N). Workgroup tile 128x128, wave tile 32x64,
// K-chunk 32, double-buffered LDS; W tiles copied via async-to-LDS.
// ---------------------------------------------------------------------------
#define TK 32

__global__ __launch_bounds__(256) void gemm_bias_kernel(
    const float* __restrict__ A, const unsigned short* __restrict__ W,
    const float* __restrict__ bias, void* __restrict__ Cout,
    int M, int N, int K, int relu, int out_bf16)
{
    __shared__ unsigned short Alds[2][128 * 32];   // 2 x 8 KiB
    __shared__ unsigned short Blds[2][32 * 128];   // 2 x 8 KiB
    const int t    = threadIdx.x;
    const int wave = t >> 5;
    const int lane = t & 31;
    const int wm   = (wave & 3) * 32;    // wave M offset in tile
    const int wn   = (wave >> 2) * 64;   // wave N offset in tile
    const int m0   = blockIdx.y * 128;
    const int n0   = blockIdx.x * 128;

    // staging coordinates (per thread, whole workgroup covers the tile)
    const int ar = t >> 1, ac = (t & 1) * 16;   // A: 16 f32 -> bf16, row ar
    const int br = t >> 3, bc = (t & 7) * 16;   // B: 16 bf16 raw copy, row br

    auto stageA = [&](int k0, int buf) {
        const float* src = &A[(size_t)(m0 + ar) * K + k0 + ac];
        unsigned short* dst = &Alds[buf][ar * 32 + ac];
#pragma unroll
        for (int i = 0; i < 16; ++i) dst[i] = f32_to_bf16(src[i]);
    };
    auto stageB = [&](int k0, int buf) {
        const unsigned short* src = &W[(size_t)(k0 + br) * N + n0 + bc];
        unsigned short* dst = &Blds[buf][br * 128 + bc];
        cp_g2l_b128(src, dst);
        cp_g2l_b128(src + 8, dst + 8);
    };

    stageA(0, 0);
    stageB(0, 0);
    async_wait();
    __syncthreads();

    v8f acc[2][4] = {};
    int buf = 0;
    for (int k0 = 0; k0 < K; k0 += TK) {
        if (k0 + TK < K) {                   // pipeline next tile into buf^1
            stageA(k0 + TK, buf ^ 1);
            stageB(k0 + TK, buf ^ 1);
        }
        if (k0 + 2 * TK < K) {               // hint tile-after-next toward L2
            __builtin_prefetch(&A[(size_t)(m0 + ar) * K + k0 + 2 * TK + ac], 0, 1);
            __builtin_prefetch(&W[(size_t)(k0 + 2 * TK + br) * N + n0 + bc], 0, 1);
        }

        v16bf a0 = load_a_frag(&Alds[buf][wm * 32], 32);
        v16bf a1 = load_a_frag(&Alds[buf][(wm + 16) * 32], 32);
#pragma unroll
        for (int j = 0; j < 4; ++j) {
            v16bf b = load_b_frag(&Blds[buf][wn + j * 16], 128);
            acc[0][j] = WMMA_BF16(a0, b, acc[0][j]);
            acc[1][j] = WMMA_BF16(a1, b, acc[1][j]);
        }
        async_wait();
        __syncthreads();
        buf ^= 1;
    }

    // epilogue: C layout VGPR r / lanes 0-15 -> M=r, lanes 16-31 -> M=r+8
    const int col0 = n0 + wn + (lane & 15);
#pragma unroll
    for (int s = 0; s < 2; ++s) {
        const int row0 = m0 + wm + 16 * s + ((lane >> 4) << 3);
#pragma unroll
        for (int j = 0; j < 4; ++j) {
            int col = col0 + j * 16;
            float bv = bias ? bias[col] : 0.f;
#pragma unroll
            for (int r = 0; r < 8; ++r) {
                float v = acc[s][j][r] + bv;
                if (relu) v = fmaxf(v, 0.f);
                size_t idx = (size_t)(row0 + r) * N + col;
                if (out_bf16) ((unsigned short*)Cout)[idx] = f32_to_bf16(v);
                else          ((float*)Cout)[idx] = v;
            }
        }
    }
}

// ---------------------------------------------------------------------------
// Fused flash attention. Q/K/V arrive as bf16 [B*S, 1024] (head-concat cols).
// grid (S/64, H, B), 128 threads = 4 waves, each wave owns a 16-row q tile.
// K/V streamed in 32-row chunks, double-buffered, async-copied to LDS.
// ---------------------------------------------------------------------------
__global__ __launch_bounds__(128) void attention_kernel(
    const unsigned short* __restrict__ Q, const unsigned short* __restrict__ Kmat,
    const unsigned short* __restrict__ V, const int* __restrict__ mask,
    float* __restrict__ ctx)
{
    const int S = 1024, D = 1024, DK = 64;
    __shared__ unsigned short Qlds[64][64];         // 8 KiB
    __shared__ unsigned short Klds[2][32][64];      // 8 KiB
    __shared__ unsigned short Vlds[2][32][64];      // 8 KiB
    __shared__ unsigned short Plds[4][16][32];      // 4 KiB (per-wave P staging)

    const int t    = threadIdx.x;
    const int wave = t >> 5;
    const int lane = t & 31;
    const int b    = blockIdx.z;
    const int h    = blockIdx.y;
    const int q0   = blockIdx.x * 64;
    const size_t base = (size_t)b * S * D + (size_t)h * DK;   // row stride D

    // stage Q tile 64x64 bf16: 32 elems / thread = 4x b128
    {
        const int qr = t >> 1, qc = (t & 1) * 32;
        const unsigned short* src = &Q[base + (size_t)(q0 + qr) * D + qc];
#pragma unroll
        for (int i = 0; i < 4; ++i) cp_g2l_b128(src + i * 8, &Qlds[qr][qc + i * 8]);
    }

    // K/V chunk staging: 16 elems / thread each = 2x b128 each
    const int kr = t >> 2, kc = (t & 3) * 16;
    auto stageKV = [&](int k0, int buf) {
        const unsigned short* ks = &Kmat[base + (size_t)(k0 + kr) * D + kc];
        const unsigned short* vs = &V   [base + (size_t)(k0 + kr) * D + kc];
        cp_g2l_b128(ks,     &Klds[buf][kr][kc]);
        cp_g2l_b128(ks + 8, &Klds[buf][kr][kc + 8]);
        cp_g2l_b128(vs,     &Vlds[buf][kr][kc]);
        cp_g2l_b128(vs + 8, &Vlds[buf][kr][kc + 8]);
    };

    stageKV(0, 0);
    async_wait();
    __syncthreads();

    v16bf aq0 = load_a_frag(&Qlds[wave * 16][0],  64);   // d = 0..31
    v16bf aq1 = load_a_frag(&Qlds[wave * 16][32], 64);   // d = 32..63

    float mrun[8], lrun[8];
#pragma unroll
    for (int r = 0; r < 8; ++r) { mrun[r] = -1e30f; lrun[r] = 0.f; }
    v8f acc[4] = {};

    int buf = 0;
    for (int k0 = 0; k0 < S; k0 += 32) {
        if (k0 + 32 < S) stageKV(k0 + 32, buf ^ 1);

        // scores s(16q x 32k) = Q(16x64) @ K^T(64x32): two 16x16 C fragments
        v16bf bk00 = load_bT_frag(&Klds[buf][0][0],   64);
        v16bf bk01 = load_bT_frag(&Klds[buf][0][32],  64);
        v16bf bk10 = load_bT_frag(&Klds[buf][16][0],  64);
        v16bf bk11 = load_bT_frag(&Klds[buf][16][32], 64);
        v8f s0 = {}, s1 = {};
        s0 = WMMA_BF16(aq0, bk00, s0);
        s0 = WMMA_BF16(aq1, bk01, s0);
        s1 = WMMA_BF16(aq0, bk10, s1);
        s1 = WMMA_BF16(aq1, bk11, s1);

        // scale 1/sqrt(64) + mask (column = k0 + 16j + lane%16)
        const bool mk0 = mask[(size_t)b * S + k0 +      (lane & 15)] != 0;
        const bool mk1 = mask[(size_t)b * S + k0 + 16 + (lane & 15)] != 0;
#pragma unroll
        for (int r = 0; r < 8; ++r) {
            s0[r] = mk0 ? s0[r] * 0.125f : -1e9f;
            s1[r] = mk1 ? s1[r] * 0.125f : -1e9f;
        }

        // online softmax; row values live across a 16-lane half-group
        float rmax[8];
#pragma unroll
        for (int r = 0; r < 8; ++r) rmax[r] = fmaxf(s0[r], s1[r]);
#pragma unroll
        for (int off = 1; off < 16; off <<= 1)
#pragma unroll
            for (int r = 0; r < 8; ++r)
                rmax[r] = fmaxf(rmax[r], __shfl_xor(rmax[r], off, 32));

        float scl[8], rsum[8];
#pragma unroll
        for (int r = 0; r < 8; ++r) {
            float mn = fmaxf(mrun[r], rmax[r]);
            scl[r]  = __expf(mrun[r] - mn);
            mrun[r] = mn;
            s0[r]   = __expf(s0[r] - mn);
            s1[r]   = __expf(s1[r] - mn);
            rsum[r] = s0[r] + s1[r];
        }
#pragma unroll
        for (int off = 1; off < 16; off <<= 1)
#pragma unroll
            for (int r = 0; r < 8; ++r) rsum[r] += __shfl_xor(rsum[r], off, 32);
#pragma unroll
        for (int r = 0; r < 8; ++r) lrun[r] = lrun[r] * scl[r] + rsum[r];
#pragma unroll
        for (int j = 0; j < 4; ++j)
#pragma unroll
            for (int r = 0; r < 8; ++r) acc[j][r] *= scl[r];

        // P (C layout) -> per-wave LDS -> A layout (wave-private, no barrier)
        const int pq = (lane >> 4) << 3;
        const int pc = lane & 15;
#pragma unroll
        for (int r = 0; r < 8; ++r) {
            Plds[wave][r + pq][pc]      = f32_to_bf16(s0[r]);
            Plds[wave][r + pq][16 + pc] = f32_to_bf16(s1[r]);
        }
        v16bf ap = load_a_frag(&Plds[wave][0][0], 32);
#pragma unroll
        for (int j = 0; j < 4; ++j) {
            v16bf bv = load_b_frag(&Vlds[buf][0][j * 16], 64);
            acc[j] = WMMA_BF16(ap, bv, acc[j]);
        }

        async_wait();
        __syncthreads();
        buf ^= 1;
    }

    // normalize and write ctx[b, s, h*64 + d] (f32, feeds O-projection GEMM)
    float inv[8];
#pragma unroll
    for (int r = 0; r < 8; ++r) inv[r] = 1.f / lrun[r];
    const int row0 = q0 + wave * 16 + ((lane >> 4) << 3);
    const int col0 = h * DK + (lane & 15);
#pragma unroll
    for (int j = 0; j < 4; ++j)
#pragma unroll
        for (int r = 0; r < 8; ++r)
            ctx[((size_t)b * S + row0 + r) * D + col0 + j * 16] = acc[j][r] * inv[r];
}

// ---------------------------------------------------------------------------
// out = LayerNorm(a + b) * gamma + beta ; one 256-thread block per row (D=1024)
// ---------------------------------------------------------------------------
__global__ __launch_bounds__(256) void add_ln_kernel(
    const float* __restrict__ a, const float* __restrict__ b,
    const float* __restrict__ gamma, const float* __restrict__ beta,
    float* __restrict__ out)
{
    const int D = 1024;
    __shared__ float red[2][8];
    const size_t row = blockIdx.x;
    const int t = threadIdx.x, wave = t >> 5, lane = t & 31;

    float v[4], s = 0.f, ss = 0.f;
#pragma unroll
    for (int i = 0; i < 4; ++i) {
        float x = a[row * D + t * 4 + i] + b[row * D + t * 4 + i];
        v[i] = x; s += x; ss += x * x;
    }
#pragma unroll
    for (int off = 1; off < 32; off <<= 1) {
        s  += __shfl_xor(s,  off, 32);
        ss += __shfl_xor(ss, off, 32);
    }
    if (lane == 0) { red[0][wave] = s; red[1][wave] = ss; }
    __syncthreads();
    float ts = 0.f, tss = 0.f;
#pragma unroll
    for (int w = 0; w < 8; ++w) { ts += red[0][w]; tss += red[1][w]; }
    const float mu = ts * (1.f / D);
    const float var = tss * (1.f / D) - mu * mu;
    const float rs = rsqrtf(var + 1e-5f);
#pragma unroll
    for (int i = 0; i < 4; ++i) {
        int c = t * 4 + i;
        out[row * D + c] = (v[i] - mu) * rs * gamma[c] + beta[c];
    }
}

// ---------------------------------------------------------------------------
// host side
// ---------------------------------------------------------------------------
extern "C" void kernel_launch(void* const* d_in, const int* in_sizes, int n_in,
                              void* d_out, int out_size, void* d_ws, size_t ws_size,
                              hipStream_t stream) {
    (void)in_sizes; (void)n_in; (void)out_size; (void)ws_size;
    const int B = 4, S = 1024, D = 1024, DFF = 4096;
    const int M = B * S;

    const float* x    = (const float*)d_in[0];
    const int*   mask = (const int*)  d_in[1];
    const float* Wq   = (const float*)d_in[2];
    const float* bq   = (const float*)d_in[3];
    const float* Wk   = (const float*)d_in[4];
    const float* bk   = (const float*)d_in[5];
    const float* Wv   = (const float*)d_in[6];
    const float* bv   = (const float*)d_in[7];
    const float* Wo   = (const float*)d_in[8];
    const float* bo   = (const float*)d_in[9];
    const float* W1   = (const float*)d_in[10];
    const float* b1   = (const float*)d_in[11];
    const float* W2   = (const float*)d_in[12];
    const float* b2   = (const float*)d_in[13];
    const float* ln1g = (const float*)d_in[14];
    const float* ln1b = (const float*)d_in[15];
    const float* ln2g = (const float*)d_in[16];
    const float* ln2b = (const float*)d_in[17];
    float* out = (float*)d_out;

    // workspace layout (176 MiB total)
    char* ws = (char*)d_ws;
    const size_t MiB = 1ull << 20;
    unsigned short* wq_bf = (unsigned short*)(ws + 0 * MiB);    // 2 MiB each
    unsigned short* wk_bf = (unsigned short*)(ws + 2 * MiB);
    unsigned short* wv_bf = (unsigned short*)(ws + 4 * MiB);
    unsigned short* wo_bf = (unsigned short*)(ws + 6 * MiB);
    unsigned short* w1_bf = (unsigned short*)(ws + 8 * MiB);    // 8 MiB
    unsigned short* w2_bf = (unsigned short*)(ws + 16 * MiB);   // 8 MiB
    unsigned short* Qb = (unsigned short*)(ws + 24 * MiB);      // 8 MiB each (bf16)
    unsigned short* Kb = (unsigned short*)(ws + 32 * MiB);
    unsigned short* Vb = (unsigned short*)(ws + 40 * MiB);
    float* ctx  = (float*)(ws + 48  * MiB);   // 16 MiB each
    float* attn = (float*)(ws + 64  * MiB);
    float* hbuf = (float*)(ws + 80  * MiB);
    float* ff1  = (float*)(ws + 96  * MiB);   // 64 MiB
    float* ffo  = (float*)(ws + 160 * MiB);   // 16 MiB

    // 1) weights -> bf16
    const int nW = D * D, nW1 = D * DFF;
    cvt_bf16_kernel<<<nW  / 256, 256, 0, stream>>>(Wq, wq_bf, nW);
    cvt_bf16_kernel<<<nW  / 256, 256, 0, stream>>>(Wk, wk_bf, nW);
    cvt_bf16_kernel<<<nW  / 256, 256, 0, stream>>>(Wv, wv_bf, nW);
    cvt_bf16_kernel<<<nW  / 256, 256, 0, stream>>>(Wo, wo_bf, nW);
    cvt_bf16_kernel<<<nW1 / 256, 256, 0, stream>>>(W1, w1_bf, nW1);
    cvt_bf16_kernel<<<nW1 / 256, 256, 0, stream>>>(W2, w2_bf, nW1);

    // 2) QKV projections (WMMA, bf16 outputs feed attention async copies)
    dim3 gP(D / 128, M / 128);
    gemm_bias_kernel<<<gP, 256, 0, stream>>>(x, wq_bf, bq, Qb, M, D, D, 0, 1);
    gemm_bias_kernel<<<gP, 256, 0, stream>>>(x, wk_bf, bk, Kb, M, D, D, 0, 1);
    gemm_bias_kernel<<<gP, 256, 0, stream>>>(x, wv_bf, bv, Vb, M, D, D, 0, 1);

    // 3) fused flash attention (WMMA for QK^T and PV)
    attention_kernel<<<dim3(S / 64, 16, B), 128, 0, stream>>>(Qb, Kb, Vb, mask, ctx);

    // 4) output projection + residual LN1
    gemm_bias_kernel<<<gP, 256, 0, stream>>>(ctx, wo_bf, bo, attn, M, D, D, 0, 0);
    add_ln_kernel<<<M, 256, 0, stream>>>(x, attn, ln1g, ln1b, hbuf);

    // 5) FFN (WMMA) + residual LN2
    dim3 gF1(DFF / 128, M / 128);
    gemm_bias_kernel<<<gF1, 256, 0, stream>>>(hbuf, w1_bf, b1, ff1, M, DFF, D, 1, 0);
    gemm_bias_kernel<<<gP,  256, 0, stream>>>(ff1, w2_bf, b2, ffo, M, D, DFF, 0, 0);
    add_ln_kernel<<<M, 256, 0, stream>>>(hbuf, ffo, ln2g, ln2b, out);
}
